// LSTMLetterPredictor_44693429682648
// MI455X (gfx1250) — matrix operand order repaired and checked by reading, not compiled, not run on hardware
//
#include <hip/hip_runtime.h>
#include <hip/hip_bf16.h>

typedef _Float16 v16h __attribute__((ext_vector_type(16)));
typedef _Float16 v8h  __attribute__((ext_vector_type(8)));
typedef _Float16 v2h  __attribute__((ext_vector_type(2)));
typedef float    v8f  __attribute__((ext_vector_type(8)));

#define B_TOTAL 16384
#define T_STEPS 32
#define I_IN    27
#define H       32
#define XS      80   // half-stride of per-wave staging rows (16B aligned)
#define NWAVES  4    // waves per block
#define BLOCK_THREADS (NWAVES * 32)
#define ROWS_PER_BLOCK (NWAVES * 16)

__device__ __forceinline__ float fsig(float x)  { return 1.0f / (1.0f + __expf(-x)); }
__device__ __forceinline__ float ftanh(float x) { return 2.0f * fsig(2.0f * x) - 1.0f; }

// A fragment (16x32 f16, M = lane&15): base = kt*32 + 8*(lane>>4); halfs [base..base+7] ++ [base+16..base+23]
__device__ __forceinline__ v16h load_a_frag(const _Float16* __restrict__ row, int kt, int lane) {
  int base = kt * 32 + 8 * (lane >> 4);
  v8h lo = *(const v8h*)(row + base);
  v8h hi = *(const v8h*)(row + base + 16);
  return __builtin_shufflevector(lo, hi, 0,1,2,3,4,5,6,7,8,9,10,11,12,13,14,15);
}

// B fragment: pre-packed per lane, 16 contiguous halfs (32B)
__device__ __forceinline__ v16h load_w_frag(const _Float16* __restrict__ w, int l, int kt, int nt, int lane) {
  const _Float16* p = w + (((((l * 2 + kt) * 8) + nt) * 32) + lane) * 16;
  v8h lo = *(const v8h*)p;
  v8h hi = *(const v8h*)(p + 8);
  return __builtin_shufflevector(lo, hi, 0,1,2,3,4,5,6,7,8,9,10,11,12,13,14,15);
}

// One layer's gate GEMM: G[16,128] = A[16,64] @ W_l[64,128], software-pipelined B-fragment stream.
__device__ __forceinline__ void gemm_layer(const _Float16* __restrict__ wlds, int l, int lane,
                                           v16h a0, v16h a1, v8f g[8]) {
  v16h nb0 = load_w_frag(wlds, l, 0, 0, lane);
  v16h nb1 = load_w_frag(wlds, l, 1, 0, lane);
  #pragma unroll
  for (int nt = 0; nt < 8; ++nt) {
    v16h cb0 = nb0, cb1 = nb1;
    if (nt < 7) {
      nb0 = load_w_frag(wlds, l, 0, nt + 1, lane);   // prefetch next tile's fragments:
      nb1 = load_w_frag(wlds, l, 1, nt + 1, lane);   // live across the WMMAs below
    }
    __builtin_amdgcn_sched_barrier(0);               // keep prefetch loads above the WMMAs
    v8f zc = {};
    v8f acc = __builtin_amdgcn_wmma_f32_16x16x32_f16(false, a0, false, cb0, (short)0, zc, false, false);
    acc = __builtin_amdgcn_wmma_f32_16x16x32_f16(false, a1, false, cb1, (short)0, acc, false, false);
    g[nt] = acc;
  }
}

__global__ __launch_bounds__(BLOCK_THREADS)
void lstm_letter_kernel(const float* __restrict__ X,      // [T*B*27] after view(T,B,-1)
                        const float* __restrict__ guessed,// [B,26]
                        const float* __restrict__ h0, const float* __restrict__ c0, // [B,32]
                        const float* __restrict__ Wih0, const float* __restrict__ Whh0,
                        const float* __restrict__ bih0, const float* __restrict__ bhh0,
                        const float* __restrict__ Wih1, const float* __restrict__ Whh1,
                        const float* __restrict__ bih1, const float* __restrict__ bhh1,
                        const float* __restrict__ Wfc, const float* __restrict__ bfc,
                        float* __restrict__ out)          // [B*26 logits | B*26 probs]
{
  __shared__ __align__(16) _Float16 wlds[2 * 2 * 8 * 32 * 16]; // 32 KB: B fragments, [l][kt][nt][lane][16]
  __shared__ float blds[2][128];                               // fused biases
  __shared__ __align__(16) _Float16 bufA[NWAVES][16 * XS];     // per wave: [x | h1] (layer 0 input)
  __shared__ __align__(16) _Float16 bufB[NWAVES][16 * XS];     // per wave: [h1 | h2] (layer 1 input)

  const int tid  = threadIdx.x;
  const int wv   = tid >> 5;
  const int lane = tid & 31;
  const int row0 = blockIdx.x * ROWS_PER_BLOCK + wv * 16;  // 16 batch rows per wave
  const int mBase = 8 * (lane >> 4);             // C/D fragment row base for this lane half
  const int nLoc  = lane & 15;                   // C/D fragment column (within 16)

  // ---- cooperatively pack weights into WMMA B-fragment layout ----
  for (int idx = tid; idx < 16384; idx += BLOCK_THREADS) {
    int p  = idx & 15;
    int ln = (idx >> 4) & 31;
    int nt = (idx >> 9) & 7;
    int kt = (idx >> 12) & 1;
    int l  = (idx >> 13) & 1;
    int K  = kt * 32 + ((ln >> 4) << 3) + (p & 7) + ((p >> 3) << 4);   // K in [0,64): [x-part | h-part]
    int N  = nt * 16 + (ln & 15);              // gate column in [0,128)
    float v;
    if (l == 0) {
      if (K < I_IN)      v = Wih0[N * I_IN + K];
      else if (K < 32)   v = 0.0f;             // pad 27->32
      else               v = Whh0[N * H + (K - 32)];
    } else {
      if (K < 32)        v = Wih1[N * H + K];
      else               v = Whh1[N * H + (K - 32)];
    }
    wlds[idx] = (_Float16)v;
  }
  for (int i = tid; i < 256; i += BLOCK_THREADS) { // fused biases
    int l = i >> 7, n = i & 127;
    blds[l][n] = (l == 0) ? (bih0[n] + bhh0[n]) : (bih1[n] + bhh1[n]);
  }
  // zero the x padding columns 27..31 of bufA (never rewritten)
  if (lane < 16) {
    for (int c = I_IN; c < 32; ++c) bufA[wv][lane * XS + c] = (_Float16)0.0f;
  }
  __syncthreads();

  // ---- load initial state as C/D fragments; init h staging buffers ----
  v8f c1[2], c2[2], hs[2];
  for (int j = 0; j < 2; ++j) {
    for (int r = 0; r < 8; ++r) {
      int M = r + mBase;
      int N = nLoc + 16 * j;
      float hv = h0[(size_t)(row0 + M) * H + N];
      float cv = c0[(size_t)(row0 + M) * H + N];
      c1[j][r] = cv;  c2[j][r] = cv;  hs[j][r] = 0.0f;   // init aliases h0/c0 for both layers
      _Float16 hh = (_Float16)hv;
      bufA[wv][M * XS + 32 + N] = hh;  // h1 init
      bufB[wv][M * XS + 32 + N] = hh;  // h2 init
    }
  }
  float biasA[8], biasB[8];              // per-lane scalar biases (added post-WMMA)
  #pragma unroll
  for (int nt = 0; nt < 8; ++nt) {
    biasA[nt] = blds[0][nt * 16 + nLoc];
    biasB[nt] = blds[1][nt * 16 + nLoc];
  }

  // ---- x staging: lane owns a contiguous run of row (lane&15) ----
  // lanes 0..15: cols [0,14); lanes 16..31: cols [14,27)
  const int xRow   = nLoc;
  const int xCol0  = (lane >> 4) * 14;
  const int xCount = (lane >> 4) ? 13 : 14;
  const int xLds   = xRow * XS + xCol0;                 // LDS half-offset (even -> b32-packable)
  const int xGlb   = xRow * I_IN + xCol0;               // offset within the wave's 432-float tile

  const _Float16* arowA = &bufA[wv][nLoc * XS]; // A-fragment row for this lane (M = lane&15)
  const _Float16* arowB = &bufB[wv][nLoc * XS];
  const size_t xstride_t = (size_t)B_TOTAL * I_IN;

  float xr[14];
  { // ---- prologue: stage x(0) ----
    const float* xsrc = X + (size_t)row0 * I_IN + xGlb;
    #pragma unroll
    for (int i = 0; i < 14; ++i) if (i < xCount) xr[i] = xsrc[i];
    #pragma unroll
    for (int i = 0; i < 7; ++i) {
      int c = 2 * i;
      if (c + 1 < xCount) {
        v2h v; v.x = (_Float16)xr[c]; v.y = (_Float16)xr[c + 1];
        *(v2h*)(&bufA[wv][xLds + c]) = v;
      } else if (c < xCount) {
        bufA[wv][xLds + c] = (_Float16)xr[c];
      }
    }
  }

  #pragma unroll 1
  for (int t = 0; t < T_STEPS; ++t) {
    asm volatile("s_wait_dscnt 0x0" ::: "memory");

    // ---- layer 0: G = [x|h1] @ W0^T (+bias in activation) ----
    v16h a0 = load_a_frag(arowA, 0, lane);
    v16h a1 = load_a_frag(arowA, 1, lane);

    // prefetch x(t+1) into registers (latency hidden behind layer-0 WMMA + activations)
    if (t + 1 < T_STEPS) {
      const float* xsrc = X + (size_t)(t + 1) * xstride_t + (size_t)row0 * I_IN + xGlb;
      #pragma unroll
      for (int i = 0; i < 14; ++i) if (i < xCount) xr[i] = xsrc[i];
      if (t + 2 < T_STEPS) {  // L2 prefetch of the tile after next (27 cachelines over 32 lanes)
        const char* pf = (const char*)(X + (size_t)(t + 2) * xstride_t + (size_t)row0 * I_IN);
        __builtin_prefetch(pf + lane * 64, 0, 0);
      }
    }

    {
      v8f g[8];
      gemm_layer(wlds, 0, lane, a0, a1, g);
      #pragma unroll
      for (int j = 0; j < 2; ++j) {
        #pragma unroll
        for (int r = 0; r < 8; ++r) {
          float iv = fsig(g[j][r] + biasA[j]);
          float fv = fsig(g[2 + j][r] + biasA[2 + j]);
          float gv = ftanh(g[4 + j][r] + biasA[4 + j]);
          float ov = fsig(g[6 + j][r] + biasA[6 + j]);
          float cn = fv * c1[j][r] + iv * gv;
          c1[j][r] = cn;
          _Float16 hh = (_Float16)(ov * ftanh(cn));
          int M = r + mBase, N = nLoc + 16 * j;
          bufB[wv][M * XS + N] = hh;        // layer-1 x-input
          bufA[wv][M * XS + 32 + N] = hh;   // next-step h1
        }
      }
    }
    asm volatile("s_wait_dscnt 0x0" ::: "memory");

    // ---- layer 1: G = [h1|h2] @ W1^T (+bias in activation) ----
    {
      v16h b_a0 = load_a_frag(arowB, 0, lane);
      v16h b_a1 = load_a_frag(arowB, 1, lane);

      // commit prefetched x(t+1): after THIS step's bufA A-loads (in-order LDS pipe),
      // and after the layer-1 A-loads so the wait above didn't cover these stores.
      if (t + 1 < T_STEPS) {
        #pragma unroll
        for (int i = 0; i < 7; ++i) {
          int c = 2 * i;
          if (c + 1 < xCount) {
            v2h v; v.x = (_Float16)xr[c]; v.y = (_Float16)xr[c + 1];
            *(v2h*)(&bufA[wv][xLds + c]) = v;
          } else if (c < xCount) {
            bufA[wv][xLds + c] = (_Float16)xr[c];
          }
        }
      }

      v8f g[8];
      gemm_layer(wlds, 1, lane, b_a0, b_a1, g);
      #pragma unroll
      for (int j = 0; j < 2; ++j) {
        #pragma unroll
        for (int r = 0; r < 8; ++r) {
          float iv = fsig(g[j][r] + biasB[j]);
          float fv = fsig(g[2 + j][r] + biasB[2 + j]);
          float gv = ftanh(g[4 + j][r] + biasB[4 + j]);
          float ov = fsig(g[6 + j][r] + biasB[6 + j]);
          float cn = fv * c2[j][r] + iv * gv;
          c2[j][r] = cn;
          float hn = ov * ftanh(cn);
          hs[j][r] += hn;                   // running sum over time of h_top
          int M = r + mBase, N = nLoc + 16 * j;
          bufB[wv][M * XS + 32 + N] = (_Float16)hn;  // next-step h2
        }
      }
    }
  }

  // ---- epilogue: havg -> LDS, then fc(58 -> 26) + sigmoid ----
  asm volatile("s_wait_dscnt 0x0" ::: "memory");
  float* havg = reinterpret_cast<float*>(&bufA[wv][0]);  // [16][32] f32 = 2 KB, fits in wave slice
  for (int j = 0; j < 2; ++j)
    for (int r = 0; r < 8; ++r) {
      int M = r + mBase, N = nLoc + 16 * j;
      havg[M * 32 + N] = hs[j][r] * (1.0f / (float)T_STEPS);
    }
  asm volatile("s_wait_dscnt 0x0" ::: "memory");

  const int rloc = nLoc;
  const int row  = row0 + rloc;
  const int obase = (lane >> 4) * 13;       // 26 outputs split 13/13 across lane halves
  const float* gs = guessed + (size_t)row * 26;
  for (int oo = 0; oo < 13; ++oo) {
    int o = obase + oo;
    float s = bfc[o];
    const float* wr = Wfc + o * (H + 26);
    #pragma unroll
    for (int k = 0; k < H; ++k)  s += wr[k] * havg[rloc * 32 + k];
    #pragma unroll
    for (int k = 0; k < 26; ++k) s += wr[H + k] * gs[k];
    out[(size_t)row * 26 + o] = s;
    out[(size_t)B_TOTAL * 26 + (size_t)row * 26 + o] = fsig(s);
  }
}

extern "C" void kernel_launch(void* const* d_in, const int* in_sizes, int n_in,
                              void* d_out, int out_size, void* d_ws, size_t ws_size,
                              hipStream_t stream) {
  const float* X    = (const float*)d_in[0];
  const float* gss  = (const float*)d_in[1];
  const float* h0   = (const float*)d_in[2];
  const float* c0   = (const float*)d_in[3];
  const float* Wih0 = (const float*)d_in[4];
  const float* Whh0 = (const float*)d_in[5];
  const float* bih0 = (const float*)d_in[6];
  const float* bhh0 = (const float*)d_in[7];
  const float* Wih1 = (const float*)d_in[8];
  const float* Whh1 = (const float*)d_in[9];
  const float* bih1 = (const float*)d_in[10];
  const float* bhh1 = (const float*)d_in[11];
  const float* Wfc  = (const float*)d_in[12];
  const float* bfc  = (const float*)d_in[13];
  float* out = (float*)d_out;

  lstm_letter_kernel<<<B_TOTAL / ROWS_PER_BLOCK, BLOCK_THREADS, 0, stream>>>(
      X, gss, h0, c0, Wih0, Whh0, bih0, bhh0, Wih1, Whh1, bih1, bhh1, Wfc, bfc, out);
}
